// CausalSelfAttention_28879360099139
// MI455X (gfx1250) — matrix operand order repaired and checked by reading, not compiled, not run on hardware
//
#include <hip/hip_runtime.h>

typedef __bf16 bf16_t;
typedef __attribute__((ext_vector_type(16))) __bf16 v16bf;
typedef __attribute__((ext_vector_type(8)))  __bf16 v8bf;
typedef __attribute__((ext_vector_type(8)))  float  v8f;

#define DIM   2048
#define T_SEQ 2048
#define NHEAD 16
#define HD    128
#define INNER 2048
#define BATCH 2
#define MROWS 4096   // B*T

// ---------- helpers ----------

// Build a 16-element bf16 fragment from two 16-byte chunks.
__device__ __forceinline__ v16bf ld2x8(const bf16_t* p0, const bf16_t* p1) {
  v8bf a = *reinterpret_cast<const v8bf*>(p0);
  v8bf b = *reinterpret_cast<const v8bf*>(p1);
  v16bf r;
#pragma unroll
  for (int i = 0; i < 8; ++i) { r[i] = a[i]; r[i + 8] = b[i]; }
  return r;
}

__device__ __forceinline__ v8f wmma_bf16(v16bf a, v16bf b, v8f c) {
  return __builtin_amdgcn_wmma_f32_16x16x32_bf16(
      /*neg_a=*/false, a, /*neg_b=*/false, b,
      /*c_mod=*/(short)0, c, /*reuse_a=*/false, /*reuse_b=*/false);
}

// ---------- conversion kernels ----------

__global__ void cvt_bf16(const float* __restrict__ in, bf16_t* __restrict__ out,
                         long long n) {
  for (long long i = (long long)blockIdx.x * blockDim.x + threadIdx.x; i < n;
       i += (long long)gridDim.x * blockDim.x)
    out[i] = (bf16_t)in[i];
}

// in[R][C] fp32 -> out[C][R] bf16
__global__ void cvt_transpose_bf16(const float* __restrict__ in,
                                   bf16_t* __restrict__ out, int R, int C) {
  const long long n = (long long)R * C;
  for (long long i = (long long)blockIdx.x * blockDim.x + threadIdx.x; i < n;
       i += (long long)gridDim.x * blockDim.x) {
    const int r = (int)(i / C);
    const int c = (int)(i % C);
    out[(size_t)c * R + r] = (bf16_t)in[i];
  }
}

// ---------- 128x128-tile bf16 WMMA GEMM ----------
// A: [M][K] bf16 row-major.  Bt: [N][K] bf16 (i.e. B transposed).
// EPI==0: QKV epilogue -> route columns to Q/K ([B,H,T,hd]) and V^T ([B,H,hd,T]).
// EPI==1: plain fp32 store to outF[M][N].
template <int EPI>
__global__ __launch_bounds__(256)
void gemm128(const bf16_t* __restrict__ A, const bf16_t* __restrict__ Bt,
             float* __restrict__ outF,
             bf16_t* __restrict__ qb, bf16_t* __restrict__ kb,
             bf16_t* __restrict__ vtb, int N, int K) {
  __shared__ bf16_t As[128 * 32];
  __shared__ bf16_t Bs[128 * 32];

  const int tid  = threadIdx.x;
  const int lane = tid & 31;
  const int w    = tid >> 5;
  const int wM   = w >> 1;   // 0..3  (32 rows each)
  const int wN   = w & 1;    // 0..1  (64 cols each)
  const int tilesN = N >> 7;
  const int m0 = ((int)blockIdx.x / tilesN) << 7;
  const int n0 = ((int)blockIdx.x % tilesN) << 7;

  const v8f zf = {};
  v8f acc[2][4];
#pragma unroll
  for (int i = 0; i < 2; ++i)
#pragma unroll
    for (int j = 0; j < 4; ++j) acc[i][j] = zf;

  const int lrow = tid >> 1;           // 0..127
  const int lcol = (tid & 1) << 4;     // 0 or 16
  const int l15  = lane & 15;
  const int kbA  = (lane >> 4) << 3;   // A fragment: chunks at kbA, kbA+16
  const int kbB  = (lane >> 4) << 3;   // B fragment: chunks at kbB, kbB+16

  for (int ks = 0; ks < K; ks += 32) {
    const bf16_t* ga = A  + (size_t)(m0 + lrow) * K + ks + lcol;
    const bf16_t* gb = Bt + (size_t)(n0 + lrow) * K + ks + lcol;
    *reinterpret_cast<v8bf*>(&As[lrow * 32 + lcol])     = *reinterpret_cast<const v8bf*>(ga);
    *reinterpret_cast<v8bf*>(&As[lrow * 32 + lcol + 8]) = *reinterpret_cast<const v8bf*>(ga + 8);
    *reinterpret_cast<v8bf*>(&Bs[lrow * 32 + lcol])     = *reinterpret_cast<const v8bf*>(gb);
    *reinterpret_cast<v8bf*>(&Bs[lrow * 32 + lcol + 8]) = *reinterpret_cast<const v8bf*>(gb + 8);
    if (ks + 32 < K) {                 // pull next k-slab toward L2/L0
      __builtin_prefetch(ga + 32, 0, 1);
      __builtin_prefetch(gb + 32, 0, 1);
    }
    __syncthreads();

    v16bf afrag[2], bfrag[4];
#pragma unroll
    for (int i = 0; i < 2; ++i) {
      const int r = (wM << 5) + (i << 4) + l15;
      afrag[i] = ld2x8(&As[r * 32 + kbA], &As[r * 32 + kbA + 16]);
    }
#pragma unroll
    for (int j = 0; j < 4; ++j) {
      const int r = (wN << 6) + (j << 4) + l15;
      bfrag[j] = ld2x8(&Bs[r * 32 + kbB], &Bs[r * 32 + kbB + 16]);
    }
#pragma unroll
    for (int i = 0; i < 2; ++i)
#pragma unroll
      for (int j = 0; j < 4; ++j)
        acc[i][j] = wmma_bf16(afrag[i], bfrag[j], acc[i][j]);
    __syncthreads();
  }

  // Epilogue. C/D layout: element r -> row M = 8*(lane>>4)+r, col N = lane&15.
  const int mL = (lane >> 4) << 3;
#pragma unroll
  for (int i = 0; i < 2; ++i) {
#pragma unroll
    for (int j = 0; j < 4; ++j) {
#pragma unroll
      for (int r = 0; r < 8; ++r) {
        const int m = m0 + (wM << 5) + (i << 4) + mL + r;
        const int n = n0 + (wN << 6) + (j << 4) + l15;
        const float val = acc[i][j][r];
        if (EPI == 0) {
          const int bb = m >> 11;          // row -> (b, t)
          const int t  = m & 2047;
          if (n < INNER) {                 // Q -> [B,H,T,hd]
            const int h = n >> 7, hd = n & 127;
            qb[(((size_t)(bb * NHEAD + h) * T_SEQ) + t) * HD + hd] = (bf16_t)val;
          } else if (n < 2 * INNER) {      // K -> [B,H,T,hd]
            const int nn = n - INNER;
            const int h = nn >> 7, hd = nn & 127;
            kb[(((size_t)(bb * NHEAD + h) * T_SEQ) + t) * HD + hd] = (bf16_t)val;
          } else {                         // V -> transposed [B,H,hd,T]
            const int nn = n - 2 * INNER;
            const int h = nn >> 7, hd = nn & 127;
            vtb[(((size_t)(bb * NHEAD + h) * HD) + hd) * T_SEQ + t] = (bf16_t)val;
          }
        } else {
          outF[(size_t)m * N + n] = val;
        }
      }
    }
  }
}

// ---------- RoPE (in place on bf16 Q/K); folds 1/sqrt(hd) into Q ----------
__global__ void rope_kernel(bf16_t* __restrict__ qb, bf16_t* __restrict__ kb) {
  const long long total = 2LL * BATCH * NHEAD * T_SEQ * 64;
  for (long long idx = (long long)blockIdx.x * blockDim.x + threadIdx.x;
       idx < total; idx += (long long)gridDim.x * blockDim.x) {
    long long i = idx;
    const int d  = (int)(i & 63);   i >>= 6;
    const int t  = (int)(i & 2047); i >>= 11;
    const int h  = (int)(i & 15);   i >>= 4;
    const int bb = (int)(i & 1);    i >>= 1;
    const bool isQ = (i == 0);
    bf16_t* p = isQ ? qb : kb;
    const float sc = isQ ? 0.08838834764831845f : 1.0f;  // 1/sqrt(128) folded into Q
    const size_t base = (((size_t)(bb * NHEAD + h) * T_SEQ) + t) * HD;
    const float x1 = (float)p[base + d];
    const float x2 = (float)p[base + d + 64];
    // inv_freq = 10000^(-d/64) = exp(-ln(10000) * d/64)
    const float ang = (float)t * __expf(-9.210340371976184f * ((float)d * 0.015625f));
    float s, c;
    __sincosf(ang, &s, &c);
    p[base + d]      = (bf16_t)(sc * (x1 * c - x2 * s));
    p[base + d + 64] = (bf16_t)(sc * (x2 * c + x1 * s));
  }
}

// ---------- flash attention: one wave == one 32-row query tile ----------
__global__ __launch_bounds__(32)
void flash_attn(const bf16_t* __restrict__ qb, const bf16_t* __restrict__ kb,
                const bf16_t* __restrict__ vtb, bf16_t* __restrict__ ctx) {
  __shared__ bf16_t P[32 * 32];       // P tile bounce buffer (C-layout -> A-layout)
  const int lane = threadIdx.x;
  const int l15  = lane & 15;
  const int half = lane >> 4;

  int g = blockIdx.x;                 // 0 .. B*H*(T/32)-1
  const int qt = g & 63; g >>= 6;
  const int h  = g & 15; g >>= 4;
  const int bb = g;
  const int q0 = qt << 5;

  const size_t hbase = (size_t)(bb * NHEAD + h) * T_SEQ * HD;
  const bf16_t* Q  = qb  + hbase;     // [t][hd]   (pre-scaled by 1/sqrt(hd))
  const bf16_t* Km = kb  + hbase;     // [t][hd]
  const bf16_t* Vt = vtb + hbase;     // [hd][t]

  // Q A-fragments: 2 row sub-tiles x 4 k-steps of the 128-deep contraction.
  v16bf qf[2][4];
#pragma unroll
  for (int i = 0; i < 2; ++i)
#pragma unroll
    for (int s = 0; s < 4; ++s) {
      const bf16_t* p =
          Q + (size_t)(q0 + (i << 4) + l15) * HD + (s << 5) + (half << 3);
      qf[i][s] = ld2x8(p, p + 16);
    }

  const v8f zf = {};
  v8f o[2][8];
#pragma unroll
  for (int i = 0; i < 2; ++i)
#pragma unroll
    for (int nt = 0; nt < 8; ++nt) o[i][nt] = zf;
  float mi[2][8], li[2][8];
#pragma unroll
  for (int i = 0; i < 2; ++i)
#pragma unroll
    for (int r = 0; r < 8; ++r) { mi[i][r] = -3.0e38f; li[i][r] = 0.0f; }

  // Online-softmax update for one 16-row sub-tile against 32 keys.
  // Lane holds key-column l15 of rows M=8*half+r; a row lives in one 16-lane
  // half, so xor-shuffles 1,2,4,8 reduce it.
  auto softmax_update = [&](int i, int k0, bool masked, v8f& sA, v8f& sB,
                            float* alpha, float* p0, float* p1) {
    const int rowbase = q0 + (i << 4) + (half << 3);
#pragma unroll
    for (int r = 0; r < 8; ++r) {
      float v0 = sA[r], v1 = sB[r];
      if (masked) {
        const int qrow = rowbase + r;
        v0 = (k0 + l15      <= qrow) ? v0 : -3.0e38f;
        v1 = (k0 + 16 + l15 <= qrow) ? v1 : -3.0e38f;
      }
      float tmax = fmaxf(v0, v1);
      tmax = fmaxf(tmax, __shfl_xor(tmax, 1));
      tmax = fmaxf(tmax, __shfl_xor(tmax, 2));
      tmax = fmaxf(tmax, __shfl_xor(tmax, 4));
      tmax = fmaxf(tmax, __shfl_xor(tmax, 8));
      const float mnew = fmaxf(mi[i][r], tmax);
      alpha[r] = __expf(mi[i][r] - mnew);
      p0[r] = __expf(v0 - mnew);
      p1[r] = __expf(v1 - mnew);
      float rs = p0[r] + p1[r];
      rs += __shfl_xor(rs, 1);
      rs += __shfl_xor(rs, 2);
      rs += __shfl_xor(rs, 4);
      rs += __shfl_xor(rs, 8);
      li[i][r] = li[i][r] * alpha[r] + rs;
      mi[i][r] = mnew;
    }
  };

  // One 32-key tile: S via 16 WMMA (K and Q fragments shared by both row
  // tiles), softmax, P bounce through LDS, PV via 16 WMMA with V fragments
  // shared by both row tiles.
  auto tile_body = [&](int k0, bool masked) {
    v8f s00 = zf, s01 = zf, s10 = zf, s11 = zf;
#pragma unroll
    for (int s = 0; s < 4; ++s) {
      const bf16_t* pk0 = Km + (size_t)(k0 + l15)      * HD + (s << 5) + (half << 3);
      const bf16_t* pk1 = Km + (size_t)(k0 + 16 + l15) * HD + (s << 5) + (half << 3);
      const v16bf kf0 = ld2x8(pk0, pk0 + 16);
      const v16bf kf1 = ld2x8(pk1, pk1 + 16);
      s00 = wmma_bf16(qf[0][s], kf0, s00);
      s01 = wmma_bf16(qf[0][s], kf1, s01);
      s10 = wmma_bf16(qf[1][s], kf0, s10);
      s11 = wmma_bf16(qf[1][s], kf1, s11);
    }

    float alpha0[8], p00[8], p01[8];
    float alpha1[8], p10[8], p11[8];
    softmax_update(0, k0, masked, s00, s01, alpha0, p00, p01);
    softmax_update(1, k0, masked, s10, s11, alpha1, p10, p11);

    // C-layout P -> LDS [row][key] -> A-layout fragments.
#pragma unroll
    for (int r = 0; r < 8; ++r) {
      P[((half << 3) + r) * 32 + l15]             = (bf16_t)p00[r];
      P[((half << 3) + r) * 32 + 16 + l15]        = (bf16_t)p01[r];
      P[(16 + (half << 3) + r) * 32 + l15]        = (bf16_t)p10[r];
      P[(16 + (half << 3) + r) * 32 + 16 + l15]   = (bf16_t)p11[r];
    }
    __syncthreads();                  // single-wave workgroup: ordering fence
    const bf16_t* pp0 = &P[l15 * 32 + (half << 3)];
    const bf16_t* pp1 = &P[(16 + l15) * 32 + (half << 3)];
    const v16bf pf0 = ld2x8(pp0, pp0 + 16);
    const v16bf pf1 = ld2x8(pp1, pp1 + 16);

    // O = O*alpha + P V  (V^T layout gives contiguous keys per hd column;
    // each V fragment feeds both row tiles).
#pragma unroll
    for (int nt = 0; nt < 8; ++nt) {
#pragma unroll
      for (int r = 0; r < 8; ++r) {
        o[0][nt][r] *= alpha0[r];
        o[1][nt][r] *= alpha1[r];
      }
      const bf16_t* pv =
          Vt + (size_t)((nt << 4) + l15) * T_SEQ + k0 + (half << 3);
      const v16bf vf = ld2x8(pv, pv + 16);
      o[0][nt] = wmma_bf16(pf0, vf, o[0][nt]);
      o[1][nt] = wmma_bf16(pf1, vf, o[1][nt]);
    }
    __syncthreads();
  };

  // All tiles strictly below the diagonal need no mask; exactly one masked
  // diagonal tile remains.
  const int nfull = q0 >> 5;
  for (int kt = 0; kt < nfull; ++kt) tile_body(kt << 5, false);
  tile_body(q0, true);

  // Normalize and write context as bf16 [B*T][INNER] for the out-projection.
#pragma unroll
  for (int i = 0; i < 2; ++i) {
    float inv[8];
#pragma unroll
    for (int r = 0; r < 8; ++r) inv[r] = 1.0f / li[i][r];
#pragma unroll
    for (int nt = 0; nt < 8; ++nt) {
#pragma unroll
      for (int r = 0; r < 8; ++r) {
        const int t   = q0 + (i << 4) + (half << 3) + r;
        const int col = h * HD + (nt << 4) + l15;
        ctx[((size_t)(bb * T_SEQ) + t) * INNER + col] =
            (bf16_t)(o[i][nt][r] * inv[r]);
      }
    }
  }
}

// ---------- host launcher ----------
extern "C" void kernel_launch(void* const* d_in, const int* in_sizes, int n_in,
                              void* d_out, int out_size, void* d_ws, size_t ws_size,
                              hipStream_t stream) {
  const float* x     = (const float*)d_in[0];   // [B,T,DIM] fp32
  const float* w_qkv = (const float*)d_in[1];   // [DIM, 3*INNER] fp32
  const float* w_out = (const float*)d_in[2];   // [INNER, DIM] fp32
  float* out = (float*)d_out;                   // [B,T,DIM] fp32

  char* ws = (char*)d_ws;
  size_t off = 0;
  bf16_t* x_bf  = (bf16_t*)(ws + off); off += (size_t)MROWS * DIM * 2;        // 16 MB
  bf16_t* wqkvT = (bf16_t*)(ws + off); off += (size_t)3 * INNER * DIM * 2;    // 24 MB
  bf16_t* woutT = (bf16_t*)(ws + off); off += (size_t)DIM * INNER * 2;        //  8 MB
  bf16_t* qb    = (bf16_t*)(ws + off); off += (size_t)MROWS * INNER * 2;      // 16 MB
  bf16_t* kb    = (bf16_t*)(ws + off); off += (size_t)MROWS * INNER * 2;      // 16 MB
  bf16_t* vtb   = (bf16_t*)(ws + off); off += (size_t)MROWS * INNER * 2;      // 16 MB
  bf16_t* ctx   = (bf16_t*)(ws + off); off += (size_t)MROWS * INNER * 2;      // 16 MB
  (void)ws_size; (void)in_sizes; (void)n_in; (void)out_size;

  cvt_bf16<<<4096, 256, 0, stream>>>(x, x_bf, (long long)MROWS * DIM);
  cvt_transpose_bf16<<<8192, 256, 0, stream>>>(w_qkv, wqkvT, DIM, 3 * INNER);
  cvt_transpose_bf16<<<4096, 256, 0, stream>>>(w_out, woutT, INNER, DIM);

  // QKV projection + routing epilogue: (4096/128)*(6144/128) = 1536 blocks
  gemm128<0><<<(MROWS / 128) * ((3 * INNER) / 128), 256, 0, stream>>>(
      x_bf, wqkvT, nullptr, qb, kb, vtb, 3 * INNER, DIM);

  rope_kernel<<<8192, 256, 0, stream>>>(qb, kb);

  // One wave per 32-row query tile: B*H*(T/32) = 2048 blocks of 32 threads.
  flash_attn<<<BATCH * NHEAD * (T_SEQ / 32), 32, 0, stream>>>(qb, kb, vtb, ctx);

  // Output projection: (4096/128)*(2048/128) = 512 blocks, fp32 out.
  gemm128<1><<<(MROWS / 128) * (DIM / 128), 256, 0, stream>>>(
      ctx, woutT, out, nullptr, nullptr, nullptr, DIM, INNER);
}
